// TriangleAttention_5789615915156
// MI455X (gfx1250) — matrix operand order repaired and checked by reading, not compiled, not run on hardware
//
#include <hip/hip_runtime.h>
#include <stdint.h>

#define Bb 2
#define Ss 256
#define Cc 128
#define Hh 4
#define Dd 32
#define Rr (Bb*Ss)

typedef __bf16 bf16_t;
typedef bf16_t v16bf __attribute__((ext_vector_type(16)));
typedef bf16_t v2bf  __attribute__((ext_vector_type(2)));
typedef float  v8f   __attribute__((ext_vector_type(8)));

union Frag { uint32_t u[8]; uint4 q4[2]; v16bf v; };

__device__ __forceinline__ uint32_t pkbf(float a, float b) {
  v2bf p; p.x = (bf16_t)a; p.y = (bf16_t)b;   // v_cvt_pk_bf16_f32
  return __builtin_bit_cast(uint32_t, p);
}
__device__ __forceinline__ float bf2f(uint32_t h) {
  return __uint_as_float(h << 16);
}
__device__ __forceinline__ v8f wmma_bf16(const Frag& a, const Frag& b, v8f c) {
  return __builtin_amdgcn_wmma_f32_16x16x32_bf16(false, a.v, false, b.v,
                                                 (short)0, c, false, false);
}

// LDS layout (bytes)
#define OFF_ZN   0          // zn bf16 [256][128]  (reused as O after projections)
#define OFF_Q    65536      // q  bf16 [256][128]  (pre-scaled by 1/sqrt(D), +bq)
#define OFF_K    131072     // k  bf16 [256][128]
#define OFF_VT   196608     // vT bf16 [128][256]  (transposed v)
#define OFF_BIAS 262144     // f32 [4][256]  bias head + key-mask (-1e4)
#define OFF_MSK  266240     // f32 [256]     query/output mask as 0/1
#define SMEM_BYTES 267264

// bf16 weight scratch layout in d_ws (uint16 elements)
#define WQ_OFF 0
#define WK_OFF 16384
#define WV_OFF 32768
#define WO_OFF 49152
#define WS_NEEDED 131072    // bytes

// ---- prep: convert f32 weights -> bf16 row-major in d_ws ----
__global__ void __launch_bounds__(256)
conv_weights_kernel(const float* __restrict__ Wq, const float* __restrict__ Wk,
                    const float* __restrict__ Wv, const float* __restrict__ Wo,
                    uint32_t* __restrict__ wbf32)
{
  int g = blockIdx.x * 256 + threadIdx.x;     // 0..32767, one uint32 (2 elems) each
  int which = g >> 13;                        // 8192 uint32 per 128x128 matrix
  int pair  = g & 8191;
  const float* src = (which == 0) ? Wq : (which == 1) ? Wk : (which == 2) ? Wv : Wo;
  const float2 v = *(const float2*)(src + (size_t)pair * 2);
  wbf32[g] = pkbf(v.x, v.y);
}

// A-fragment from weights (flipped GEMMs): u[0..3], u[4..7] 16B blocks 32B apart
template<bool WS>
__device__ __forceinline__ void loadWA(Frag& a, const uint16_t* w16, const float* wf,
                                       int row, int kc, int hl) {
  if (WS) {
    const uint4* p = (const uint4*)(w16 + (size_t)row * Cc + kc*32 + hl*8);
    a.q4[0] = p[0];
    a.q4[1] = p[2];
  } else {
    #pragma unroll
    for (int i = 0; i < 8; ++i) {
      int kk = kc*32 + ((i >> 2) << 4) + hl*8 + ((i & 3) << 1);
      const float2 wp = *(const float2*)(wf + (size_t)row * Cc + kk);
      a.u[i] = pkbf(wp.x, wp.y);
    }
  }
}
// B-fragment from weights (v projection): 32B contiguous
template<bool WS>
__device__ __forceinline__ void loadWB(Frag& b, const uint16_t* w16, const float* wf,
                                       int row, int kbase) {
  if (WS) {
    const uint4* p = (const uint4*)(w16 + (size_t)row * Cc + kbase);
    b.q4[0] = p[0];
    b.q4[1] = p[1];
  } else {
    #pragma unroll
    for (int i = 0; i < 8; ++i) {
      const float2 wp = *(const float2*)(wf + (size_t)row * Cc + kbase + 2*i);
      b.u[i] = pkbf(wp.x, wp.y);
    }
  }
}

template<bool USE_WS>
__global__ void __launch_bounds__(256)
tri_attn_kernel(const float* __restrict__ z, const uint8_t* __restrict__ pmask,
                const float* __restrict__ lnw, const float* __restrict__ lnb,
                const float* __restrict__ Wq, const float* __restrict__ bq,
                const float* __restrict__ Wk, const float* __restrict__ bk,
                const float* __restrict__ Wv, const float* __restrict__ bv,
                const float* __restrict__ Wb, const float* __restrict__ Wo,
                const float* __restrict__ bo, const uint16_t* __restrict__ wbf,
                float* __restrict__ out)
{
  extern __shared__ __align__(16) char smem[];
  char*  zns   = smem + OFF_ZN;
  char*  qs    = smem + OFF_Q;
  char*  ks    = smem + OFF_K;
  char*  vts   = smem + OFF_VT;
  float* biasF = (float*)(smem + OFF_BIAS);
  float* maskF = (float*)(smem + OFF_MSK);

  const int tid  = threadIdx.x;
  const int lane = tid & 31;
  const int wv   = tid >> 5;      // wave id 0..7
  const int r    = blockIdx.x;    // row 0..511
  const int m    = lane & 15;
  const int hl   = lane >> 4;

  const uint16_t* wq16 = USE_WS ? wbf + WQ_OFF : nullptr;
  const uint16_t* wk16 = USE_WS ? wbf + WK_OFF : nullptr;
  const uint16_t* wv16 = USE_WS ? wbf + WV_OFF : nullptr;
  const uint16_t* wo16 = USE_WS ? wbf + WO_OFF : nullptr;

  // ---------------- Phase 1: LayerNorm -> zn (bf16, LDS) ----------------
  {
    const float* zrow = z + (size_t)r * (Ss * Cc);
    const float4 w4 = ((const float4*)lnw)[lane];
    const float4 b4 = ((const float4*)lnb)[lane];
    for (int j = 0; j < 32; ++j) {
      int s = wv * 32 + j;
      float4 zv = ((const float4*)(zrow + s * Cc))[lane];
      float sm = zv.x + zv.y + zv.z + zv.w;
      float sq = zv.x*zv.x + zv.y*zv.y + zv.z*zv.z + zv.w*zv.w;
      #pragma unroll
      for (int d = 16; d >= 1; d >>= 1) {
        sm += __shfl_xor(sm, d, 32);
        sq += __shfl_xor(sq, d, 32);
      }
      float mu  = sm * (1.0f / 128.0f);
      float var = sq * (1.0f / 128.0f) - mu * mu;
      float rs  = rsqrtf(var + 1e-5f);
      float n0 = (zv.x - mu) * rs * w4.x + b4.x;
      float n1 = (zv.y - mu) * rs * w4.y + b4.y;
      float n2 = (zv.z - mu) * rs * w4.z + b4.z;
      float n3 = (zv.w - mu) * rs * w4.w + b4.w;
      *(uint2*)(zns + (size_t)(s * Cc + lane * 4) * 2) =
          make_uint2(pkbf(n0, n1), pkbf(n2, n3));
    }
  }
  __syncthreads();

  // ------------- Phase 2: bias head + mask table + q/k/v projections -------------
  {
    int s = tid;
    uint8_t mk = pmask[(size_t)r * Ss + s];
    maskF[s] = mk ? 1.0f : 0.0f;
    float d0 = 0.f, d1 = 0.f, d2 = 0.f, d3 = 0.f;
    const uint32_t* znr = (const uint32_t*)(zns + (size_t)s * Cc * 2);
    #pragma unroll 4
    for (int c2 = 0; c2 < 64; ++c2) {
      uint32_t p = znr[c2];
      float a = bf2f(p & 0xFFFFu), b = bf2f(p >> 16);
      int c = c2 * 2;
      d0 += a * Wb[0*Cc + c] + b * Wb[0*Cc + c + 1];
      d1 += a * Wb[1*Cc + c] + b * Wb[1*Cc + c + 1];
      d2 += a * Wb[2*Cc + c] + b * Wb[2*Cc + c + 1];
      d3 += a * Wb[3*Cc + c] + b * Wb[3*Cc + c + 1];
    }
    float madd = mk ? 0.0f : -10000.0f;
    biasF[0*Ss + s] = d0 + madd;
    biasF[1*Ss + s] = d1 + madd;
    biasF[2*Ss + s] = d2 + madd;
    biasF[3*Ss + s] = d3 + madd;
  }

  // q/k flipped (A=W, B=zn) -> contiguous b128 stores; v as-is -> b128 into vT
  // (ob = t&7 is wave-invariant -> compiler hoists all weight fragments)
  for (int t = wv; t < 128; t += 8) {
    int sb = t >> 3, ob = t & 7;
    Frag bzn[4];                       // zn B-frags shared by q and k
    #pragma unroll
    for (int kc = 0; kc < 4; ++kc) {
      const uint4* p = (const uint4*)(zns +
          (size_t)((sb*16 + m) * Cc + kc*32 + hl*8) * 2);
      bzn[kc].q4[0] = p[0];
      bzn[kc].q4[1] = p[2];
    }
    // ---- q^T tile ----
    {
      v8f acc = {0.f,0.f,0.f,0.f,0.f,0.f,0.f,0.f};
      #pragma unroll
      for (int kc = 0; kc < 4; ++kc) {
        Frag a; loadWA<USE_WS>(a, wq16, Wq, ob*16 + m, kc, hl);
        acc = wmma_bf16(a, bzn[kc], acc);
      }
      const float4* bp = (const float4*)(bq + ob*16 + hl*8);
      float4 b0 = bp[0], b1 = bp[1];
      const float sc = 0.17677669529663687f;   // 1/sqrt(32)
      uint4 st;
      st.x = pkbf((acc[0]+b0.x)*sc, (acc[1]+b0.y)*sc);
      st.y = pkbf((acc[2]+b0.z)*sc, (acc[3]+b0.w)*sc);
      st.z = pkbf((acc[4]+b1.x)*sc, (acc[5]+b1.y)*sc);
      st.w = pkbf((acc[6]+b1.z)*sc, (acc[7]+b1.w)*sc);
      *(uint4*)(qs + (size_t)((sb*16 + m) * Cc + ob*16 + hl*8) * 2) = st;
    }
    // ---- k^T tile ----
    {
      v8f acc = {0.f,0.f,0.f,0.f,0.f,0.f,0.f,0.f};
      #pragma unroll
      for (int kc = 0; kc < 4; ++kc) {
        Frag a; loadWA<USE_WS>(a, wk16, Wk, ob*16 + m, kc, hl);
        acc = wmma_bf16(a, bzn[kc], acc);
      }
      const float4* bp = (const float4*)(bk + ob*16 + hl*8);
      float4 b0 = bp[0], b1 = bp[1];
      uint4 st;
      st.x = pkbf(acc[0]+b0.x, acc[1]+b0.y);
      st.y = pkbf(acc[2]+b0.z, acc[3]+b0.w);
      st.z = pkbf(acc[4]+b1.x, acc[5]+b1.y);
      st.w = pkbf(acc[6]+b1.z, acc[7]+b1.w);
      *(uint4*)(ks + (size_t)((sb*16 + m) * Cc + ob*16 + hl*8) * 2) = st;
    }
    // ---- v tile (A=zn, B=Wv) stored transposed -> b128 into vT ----
    {
      v8f acc = {0.f,0.f,0.f,0.f,0.f,0.f,0.f,0.f};
      #pragma unroll
      for (int kc = 0; kc < 4; ++kc) {
        Frag a, b;
        const uint4* pa = (const uint4*)(zns +
            (size_t)((sb*16 + m) * Cc + kc*32 + hl*16) * 2);
        a.q4[0] = pa[0];
        a.q4[1] = pa[1];
        loadWB<USE_WS>(b, wv16, Wv, ob*16 + m, kc*32 + hl*16);
        acc = wmma_bf16(a, b, acc);
      }
      float badd = bv[ob*16 + m];
      uint4 st;
      st.x = pkbf(acc[0]+badd, acc[1]+badd);
      st.y = pkbf(acc[2]+badd, acc[3]+badd);
      st.z = pkbf(acc[4]+badd, acc[5]+badd);
      st.w = pkbf(acc[6]+badd, acc[7]+badd);
      *(uint4*)(vts + (size_t)((ob*16 + m) * Ss + sb*16 + hl*8) * 2) = st;
    }
  }
  __syncthreads();

  // ------- Phase 3: attention (transposed logits, single unit per pass) -------
  {
    const int h   = wv >> 1;              // head fixed per wave
    const int qb0 = (wv & 1) * 8;
    #pragma unroll 1
    for (int j = 0; j < 8; ++j) {
      int qb = qb0 + j;
      Frag bq_;                           // q B-frag (K = 32 = full head dim)
      {
        const uint4* p = (const uint4*)(qs +
            (size_t)((qb*16 + m) * Cc + h*Dd + hl*8) * 2);
        bq_.q4[0] = p[0];
        bq_.q4[1] = p[2];
      }
      v8f lg[16];                         // logits^T: lane holds (sq=m, t=tt*16+i+8hl)
      #pragma unroll
      for (int tt = 0; tt < 16; ++tt) {
        Frag ak;
        const uint4* p = (const uint4*)(ks +
            (size_t)((tt*16 + m) * Cc + h*Dd + hl*8) * 2);
        ak.q4[0] = p[0];
        ak.q4[1] = p[2];
        v8f c = {0.f,0.f,0.f,0.f,0.f,0.f,0.f,0.f};
        c = wmma_bf16(ak, bq_, c);
        const float4* bp = (const float4*)(biasF + h*Ss + tt*16 + hl*8);
        float4 ba = bp[0], bb = bp[1];
        c[0] += ba.x; c[1] += ba.y; c[2] += ba.z; c[3] += ba.w;
        c[4] += bb.x; c[5] += bb.y; c[6] += bb.z; c[7] += bb.w;
        lg[tt] = c;
      }
      // softmax stats: row sq=m lives in lanes (m, m+16) -> one shfl_xor(16) each.
      // 1/sum is folded into the O epilogue, not applied to the 128 attn regs.
      float mx = lg[0][0];
      #pragma unroll
      for (int tt = 0; tt < 16; ++tt)
        #pragma unroll
        for (int i = 0; i < 8; ++i) mx = fmaxf(mx, lg[tt][i]);
      mx = fmaxf(mx, __shfl_xor(mx, 16, 32));
      float smv = 0.f;
      #pragma unroll
      for (int tt = 0; tt < 16; ++tt)
        #pragma unroll
        for (int i = 0; i < 8; ++i) {
          float e = __expf(lg[tt][i] - mx);
          lg[tt][i] = e;
          smv += e;
        }
      smv += __shfl_xor(smv, 16, 32);
      float inv = 1.0f / smv;

      // O^T = v^T x attn^T ; attn B-frags built by pure register exchange
      v8f o0 = {0.f,0.f,0.f,0.f,0.f,0.f,0.f,0.f};
      v8f o1 = {0.f,0.f,0.f,0.f,0.f,0.f,0.f,0.f};
      #pragma unroll
      for (int kc = 0; kc < 8; ++kc) {
        Frag ba;
        #pragma unroll
        for (int b = 0; b < 4; ++b) {
          uint32_t p0 = pkbf(lg[kc*2    ][2*b], lg[kc*2    ][2*b+1]); // tile kc*2
          uint32_t p1 = pkbf(lg[kc*2 + 1][2*b], lg[kc*2 + 1][2*b+1]); // tile kc*2+1
          uint32_t send = hl ? p0 : p1;
          uint32_t keep = hl ? p1 : p0;
          uint32_t recv = (uint32_t)__shfl_xor((int)send, 16, 32);
          ba.u[b]     = hl ? recv : keep;
          ba.u[4 + b] = hl ? keep : recv;
        }
        Frag a0, a1;
        const uint4* pv0 = (const uint4*)(vts +
            (size_t)((h*Dd + m) * Ss + kc*32 + hl*16) * 2);
        a0.q4[0] = pv0[0];
        a0.q4[1] = pv0[1];
        const uint4* pv1 = (const uint4*)(vts +
            (size_t)((h*Dd + 16 + m) * Ss + kc*32 + hl*16) * 2);
        a1.q4[0] = pv1[0];
        a1.q4[1] = pv1[1];
        o0 = wmma_bf16(a0, ba, o0);
        o1 = wmma_bf16(a1, ba, o1);
      }
      // store O^T tiles -> row-major O (zn region); fold 1/sum here
      uint4 s0, s1;
      s0.x = pkbf(o0[0]*inv, o0[1]*inv); s0.y = pkbf(o0[2]*inv, o0[3]*inv);
      s0.z = pkbf(o0[4]*inv, o0[5]*inv); s0.w = pkbf(o0[6]*inv, o0[7]*inv);
      s1.x = pkbf(o1[0]*inv, o1[1]*inv); s1.y = pkbf(o1[2]*inv, o1[3]*inv);
      s1.z = pkbf(o1[4]*inv, o1[5]*inv); s1.w = pkbf(o1[6]*inv, o1[7]*inv);
      *(uint4*)(zns + (size_t)((qb*16 + m) * Cc + h*Dd + hl*8) * 2)      = s0;
      *(uint4*)(zns + (size_t)((qb*16 + m) * Cc + h*Dd + 16 + hl*8) * 2) = s1;
    }
  }
  __syncthreads();

  // ---------------- Phase 4: out^T = Wo x O^T, +bo, mask, b128 stores ----------------
  for (int t = wv; t < 128; t += 8) {
    int sb = t >> 3, ob = t & 7;
    v8f acc = {0.f,0.f,0.f,0.f,0.f,0.f,0.f,0.f};
    #pragma unroll
    for (int kc = 0; kc < 4; ++kc) {
      Frag a, b;
      loadWA<USE_WS>(a, wo16, Wo, ob*16 + m, kc, hl);
      const uint4* p = (const uint4*)(zns +
          (size_t)((sb*16 + m) * Cc + kc*32 + hl*8) * 2);
      b.q4[0] = p[0];
      b.q4[1] = p[2];
      acc = wmma_bf16(a, b, acc);
    }
    const float4* bp = (const float4*)(bo + ob*16 + hl*8);
    float4 b0 = bp[0], b1 = bp[1];
    float mv = maskF[sb*16 + m];
    float4 r0, r1;
    r0.x = (acc[0] + b0.x) * mv; r0.y = (acc[1] + b0.y) * mv;
    r0.z = (acc[2] + b0.z) * mv; r0.w = (acc[3] + b0.w) * mv;
    r1.x = (acc[4] + b1.x) * mv; r1.y = (acc[5] + b1.y) * mv;
    r1.z = (acc[6] + b1.z) * mv; r1.w = (acc[7] + b1.w) * mv;
    float* op = out + ((size_t)r * Ss + sb*16 + m) * Cc + ob*16 + hl*8;
    *(float4*)(op)     = r0;
    *(float4*)(op + 4) = r1;
  }
}

extern "C" void kernel_launch(void* const* d_in, const int* in_sizes, int n_in,
                              void* d_out, int out_size, void* d_ws, size_t ws_size,
                              hipStream_t stream) {
  (void)in_sizes; (void)n_in; (void)out_size;
  const float*   z   = (const float*)d_in[0];
  const uint8_t* pm  = (const uint8_t*)d_in[1];
  const float*   lnw = (const float*)d_in[2];
  const float*   lnb = (const float*)d_in[3];
  const float*   Wq  = (const float*)d_in[4];
  const float*   bq  = (const float*)d_in[5];
  const float*   Wk  = (const float*)d_in[6];
  const float*   bk  = (const float*)d_in[7];
  const float*   Wv  = (const float*)d_in[8];
  const float*   bv  = (const float*)d_in[9];
  const float*   Wb  = (const float*)d_in[10];
  const float*   Wo  = (const float*)d_in[11];
  const float*   bo  = (const float*)d_in[12];
  float* out = (float*)d_out;

  const bool use_ws = (ws_size >= (size_t)WS_NEEDED) && (d_ws != nullptr);
  if (use_ws) {
    conv_weights_kernel<<<dim3(128), dim3(256), 0, stream>>>(
        Wq, Wk, Wv, Wo, (uint32_t*)d_ws);
    (void)hipFuncSetAttribute((const void*)tri_attn_kernel<true>,
                              hipFuncAttributeMaxDynamicSharedMemorySize,
                              (int)SMEM_BYTES);
    tri_attn_kernel<true><<<dim3(Rr), dim3(256), SMEM_BYTES, stream>>>(
        z, pm, lnw, lnb, Wq, bq, Wk, bk, Wv, bv, Wb, Wo, bo,
        (const uint16_t*)d_ws, out);
  } else {
    (void)hipFuncSetAttribute((const void*)tri_attn_kernel<false>,
                              hipFuncAttributeMaxDynamicSharedMemorySize,
                              (int)SMEM_BYTES);
    tri_attn_kernel<false><<<dim3(Rr), dim3(256), SMEM_BYTES, stream>>>(
        z, pm, lnw, lnb, Wq, bq, Wk, bk, Wv, bv, Wb, Wo, bo, nullptr, out);
  }
}
